// SPVCNN_13142599926371
// MI455X (gfx1250) — compile-verified, matrix operand order checked by reading, not run on hardware
//
#include <hip/hip_runtime.h>

// ---------------------------------------------------------------------------
// SPVCNN forward on gfx1250 (CDNA5 / MI455X).
// Convs/linears -> f16 WMMA GEMMs (v_wmma_f32_16x16x32_f16) with LDS-staged
// im2col A tiles (ds_read_b128 fragment loads) and TDM (tensor_load_to_lds)
// double-buffered weight slabs in LDS, synchronized with s_wait_tensorcnt.
// Voxelize/devoxelize are bandwidth-bound scatter/gather; training-mode BN is
// a per-channel reduction + fused apply.
// ---------------------------------------------------------------------------

typedef __attribute__((ext_vector_type(16))) _Float16 v16h;
typedef __attribute__((ext_vector_type(8)))  _Float16 v8h;
typedef __attribute__((ext_vector_type(8)))  float    v8f;
typedef __attribute__((ext_vector_type(4)))  float    v4f;
typedef __attribute__((ext_vector_type(4)))  unsigned int u32x4;
typedef __attribute__((ext_vector_type(4)))  int      i32x4;
typedef __attribute__((ext_vector_type(8)))  int      i32x8;

#define BN_EPS 1e-5f

#if __has_builtin(__builtin_amdgcn_tensor_load_to_lds) && \
    __has_builtin(__builtin_amdgcn_s_wait_tensorcnt)
#define USE_TDM 1
#else
#define USE_TDM 0
#endif

__device__ __forceinline__ int iminD(int a, int b) { return a < b ? a : b; }
__device__ __forceinline__ int imaxD(int a, int b) { return a > b ? a : b; }

// ---------------------------------------------------------------------------
// Elementwise helpers
// ---------------------------------------------------------------------------
__global__ void k_zero(float* p, long n) {
    long i = (long)blockIdx.x * blockDim.x + threadIdx.x;
    for (; i < n; i += (long)gridDim.x * blockDim.x) p[i] = 0.0f;
}

__global__ void k_add_inplace(float* a, const float* b, long n) {
    long i = (long)blockIdx.x * blockDim.x + threadIdx.x;
    for (; i < n; i += (long)gridDim.x * blockDim.x) a[i] += b[i];
}

__global__ void k_concat(const float* __restrict__ a, int Ca,
                         const float* __restrict__ b, int Cb,
                         float* __restrict__ out, long M) {
    int C = Ca + Cb;
    long tot = M * C;
    long i = (long)blockIdx.x * blockDim.x + threadIdx.x;
    for (; i < tot; i += (long)gridDim.x * blockDim.x) {
        long m = i / C; int c = (int)(i % C);
        out[i] = (c < Ca) ? a[m * Ca + c] : b[m * Cb + (c - Ca)];
    }
}

// Weight f32 [K][cout] -> TRANSPOSED f16 whT[cout][Kp], K padded to Kp.
__global__ void k_w2hT(const float* __restrict__ w, _Float16* __restrict__ whT,
                       int K, int Kp, int cout) {
    long tot = (long)cout * Kp;
    long i = (long)blockIdx.x * blockDim.x + threadIdx.x;
    for (; i < tot; i += (long)gridDim.x * blockDim.x) {
        int k = (int)(i % Kp);
        int n = (int)(i / Kp);
        whT[i] = (k < K) ? (_Float16)w[(long)k * cout + n] : (_Float16)0.0f;
    }
}

// ---------------------------------------------------------------------------
// Voxelize: scatter-mean of point features into a dense grid
// ---------------------------------------------------------------------------
__global__ void k_scatter(const float* __restrict__ f, const float* __restrict__ coords,
                          float* __restrict__ sums, float* __restrict__ cnt,
                          int N, int C, float stride, int d) {
    long tot = (long)N * C;
    long i = (long)blockIdx.x * blockDim.x + threadIdx.x;
    for (; i < tot; i += (long)gridDim.x * blockDim.x) {
        int p = (int)(i / C); int c = (int)(i % C);
        int ix = iminD(imaxD((int)floorf(coords[p * 3 + 0] / stride), 0), d - 1);
        int iy = iminD(imaxD((int)floorf(coords[p * 3 + 1] / stride), 0), d - 1);
        int iz = iminD(imaxD((int)floorf(coords[p * 3 + 2] / stride), 0), d - 1);
        long v = ((long)ix * d + iy) * d + iz;
        atomicAdd(&sums[v * C + c], f[i]);
        if (c == 0) atomicAdd(&cnt[v], 1.0f);
    }
}

__global__ void k_voxnorm(float* __restrict__ g, const float* __restrict__ cnt,
                          long M, int C) {
    long tot = M * C;
    long i = (long)blockIdx.x * blockDim.x + threadIdx.x;
    for (; i < tot; i += (long)gridDim.x * blockDim.x)
        g[i] = g[i] / fmaxf(cnt[i / C], 1.0f);
}

// ---------------------------------------------------------------------------
// Devoxelize: trilinear gather of grid features at point locations
// ---------------------------------------------------------------------------
__global__ void k_devox(const float* __restrict__ grid, const float* __restrict__ coords,
                        float* __restrict__ out, int N, int C, float stride, int d) {
    long tot = (long)N * C;
    long i = (long)blockIdx.x * blockDim.x + threadIdx.x;
    for (; i < tot; i += (long)gridDim.x * blockDim.x) {
        int p = (int)(i / C); int c = (int)(i % C);
        float cx = coords[p * 3 + 0] / stride;
        float cy = coords[p * 3 + 1] / stride;
        float cz = coords[p * 3 + 2] / stride;
        float fx = floorf(cx), fy = floorf(cy), fz = floorf(cz);
        float rx = cx - fx, ry = cy - fy, rz = cz - fz;
        int ix = (int)fx, iy = (int)fy, iz = (int)fz;
        float acc = 0.0f;
#pragma unroll
        for (int dx = 0; dx < 2; ++dx)
#pragma unroll
            for (int dy = 0; dy < 2; ++dy)
#pragma unroll
                for (int dz = 0; dz < 2; ++dz) {
                    int jx = iminD(imaxD(ix + dx, 0), d - 1);
                    int jy = iminD(imaxD(iy + dy, 0), d - 1);
                    int jz = iminD(imaxD(iz + dz, 0), d - 1);
                    float w = (dx ? rx : 1.0f - rx) * (dy ? ry : 1.0f - ry) *
                              (dz ? rz : 1.0f - rz);
                    acc += w * grid[(((long)jx * d + jy) * d + jz) * C + c];
                }
        out[i] = acc;
    }
}

// ---------------------------------------------------------------------------
// Training-mode BatchNorm: per-channel mean/var, then fused apply
// ---------------------------------------------------------------------------
__global__ void k_bn_stats(const float* __restrict__ x, long M, int C,
                           float* __restrict__ mean, float* __restrict__ var) {
    int c = blockIdx.x;
    __shared__ float s1[256];
    __shared__ float s2[256];
    float a = 0.0f, b = 0.0f;
    for (long m = threadIdx.x; m < M; m += blockDim.x) {
        float v = x[m * C + c];
        a += v; b += v * v;
    }
    s1[threadIdx.x] = a; s2[threadIdx.x] = b;
    __syncthreads();
    for (int s = 128; s > 0; s >>= 1) {
        if ((int)threadIdx.x < s) {
            s1[threadIdx.x] += s1[threadIdx.x + s];
            s2[threadIdx.x] += s2[threadIdx.x + s];
        }
        __syncthreads();
    }
    if (threadIdx.x == 0) {
        float mu = s1[0] / (float)M;
        mean[c] = mu;
        var[c] = s2[0] / (float)M - mu * mu;
    }
}

__global__ void k_bn_apply(const float* __restrict__ x, const float* __restrict__ mean,
                           const float* __restrict__ var, const float* __restrict__ g,
                           const float* __restrict__ b, const float* __restrict__ res,
                           float* __restrict__ out, long M, int C, int relu) {
    long tot = M * C;
    long i = (long)blockIdx.x * blockDim.x + threadIdx.x;
    for (; i < tot; i += (long)gridDim.x * blockDim.x) {
        int c = (int)(i % C);
        float sc = g[c] * rsqrtf(var[c] + BN_EPS);
        float sh = b[c] - mean[c] * sc;
        float v = x[i] * sc + sh;
        if (res) v += res[i];
        out[i] = relu ? fmaxf(v, 0.0f) : v;
    }
}

// ---------------------------------------------------------------------------
// im2col tap -> input base offset; valid=false => zero fill.
// mode 0: conv (ks^3 taps).  mode 1: ks2/stride2 transposed conv.
// ---------------------------------------------------------------------------
__device__ __forceinline__ long conv_base(int tap, int ox, int oy, int oz,
                                          int d_in, int ks, int stride, int pad,
                                          int cin, int mode, bool& valid) {
    if (mode == 0) {
        int kz = tap % ks; int t2 = tap / ks;
        int ky = t2 % ks; int kx = t2 / ks;
        int ix = ox * stride + kx - pad;
        int iy = oy * stride + ky - pad;
        int iz = oz * stride + kz - pad;
        valid = (unsigned)ix < (unsigned)d_in && (unsigned)iy < (unsigned)d_in &&
                (unsigned)iz < (unsigned)d_in;
        return valid ? ((((long)ix * d_in + iy) * d_in + iz) * (long)cin) : 0;
    } else {
        int kz = tap & 1, ky = (tap >> 1) & 1, kx = (tap >> 2) & 1;
        valid = ((ox & 1) == kx) && ((oy & 1) == ky) && ((oz & 1) == kz);
        int ix = ox >> 1, iy = oy >> 1, iz = oz >> 1;
        return valid ? ((((long)ix * d_in + iy) * d_in + iz) * (long)cin) : 0;
    }
}

// ---------------------------------------------------------------------------
// TDM: DMA one weight slab whT[0..cout)[k0..k0+32) (2D strided tile, 2-byte
// elements, row stride Kp) into LDS at lds_off.  D# packed per ISA 8.3/8.4:
// group0 = {count=1, lds_addr, global_addr[56:0], type=2},
// group1 = {data_size=2B, tensor_dim0=Kp, tensor_dim1=cout, tile_dim0=32,
//           tile_dim1=cout, tensor_dim0_stride=Kp}.
// ---------------------------------------------------------------------------
__device__ __forceinline__ void tdm_load_b_slab(const _Float16* gsrc,
                                                unsigned lds_off, int Kp, int cout) {
#if USE_TDM
    unsigned long long ga = (unsigned long long)(const void*)gsrc;
    u32x4 g0 = {};
    g0[0] = 1u;                                   // count=1 (valid user D#)
    g0[1] = lds_off;                              // lds_addr (bytes)
    g0[2] = (unsigned)(ga & 0xFFFFFFFFu);         // global_addr[31:0]
    g0[3] = (unsigned)((ga >> 32) & 0x01FFFFFFu)  // global_addr[56:32]
            | (2u << 30);                         // type = 2 ("image")
    i32x8 g1 = {};
    g1[0] = (int)(1u << 16);                                  // data_size=1 -> 2B
    g1[1] = (int)(((unsigned)Kp & 0xFFFFu) << 16);            // tensor_dim0 lo16
    g1[2] = (int)((((unsigned)Kp >> 16) & 0xFFFFu)            // tensor_dim0 hi16
                  | (((unsigned)cout & 0xFFFFu) << 16));      // tensor_dim1 lo16
    g1[3] = (int)(32u << 16);                                 // tile_dim0 = 32
    g1[4] = (int)((unsigned)cout & 0xFFFFu);                  // tile_dim1 = cout
    g1[5] = Kp;                                               // dim0_stride lo32
    g1[6] = 0;
    g1[7] = 0;
    i32x4 g2 = {};
    i32x4 g3 = {};
#if __clang_major__ >= 23
    i32x8 g4 = {};
    __builtin_amdgcn_tensor_load_to_lds(g0, g1, g2, g3, g4, 0);
#else
    __builtin_amdgcn_tensor_load_to_lds(g0, g1, g2, g3, 0);
#endif
#else
    (void)gsrc; (void)lds_off; (void)Kp; (void)cout;
#endif
}

// ---------------------------------------------------------------------------
// Conv3D / deconv as WMMA GEMM.
// Block = 256 threads = 8 waves -> 128 output voxels x cout (NT*16 channels).
// Per K-step: cooperative LDS fill of im2col A[128][32] f16 (incremental tap
// decode), TDM-prefetched double-buffered B slab in LDS, ds_read_b128
// fragment loads, NT wmma ops into register accumulators.
// ---------------------------------------------------------------------------
#define AROWS 40

template <int NT>
__global__ __launch_bounds__(256)
void k_conv_wmma(const float* __restrict__ in, const _Float16* __restrict__ whT,
                 float* __restrict__ out, int d_in, int d_out, int cin, int cout,
                 int K, int Kp, int ks, int stride, int pad, int mode) {
    __shared__ _Float16 lds_a[128 * AROWS];
    __shared__ _Float16 lds_b[2][64 * 32];   // double-buffered weight slabs
    const int tid  = threadIdx.x;
    const int lane = tid & 31;
    const int wave = tid >> 5;
    const long Mtot = (long)d_out * d_out * d_out;
    const long mblk = (long)blockIdx.x * 128;

    // fill-phase: thread -> (row, 16-element kloc run)
    const int frow   = tid >> 1;
    const int kstart = (tid & 1) * 16;
    long fm = mblk + frow; if (fm >= Mtot) fm = Mtot - 1;
    const int foz = (int)(fm % d_out);
    const long ftq = fm / d_out;
    const int foy = (int)(ftq % d_out);
    const int fox = (int)(ftq / d_out);

    // compute-phase: ISA 7.12.2 16-bit A/B layout -> two contiguous 8-half
    // chunks per lane at kloc {khalf..+7} and {16+khalf..+7}.
    const int arow  = wave * 16 + (lane & 15);
    const int khalf = (lane >> 4) * 8;
    const int bcol  = lane & 15;

    v8f acc[NT];
#pragma unroll
    for (int nt = 0; nt < NT; ++nt) acc[nt] = (v8f){};

    const int nslab = Kp / 32;
#if USE_TDM
    if (wave == 0)   // prime the pipeline: slab 0 -> buffer 0
        tdm_load_b_slab(whT, (unsigned)(unsigned long long)(void*)&lds_b[0][0],
                        Kp, cout);
#endif

    for (int s = 0; s < nslab; ++s) {
        const int k0 = s * 32;
        // ---- cooperative LDS fill of A tile (incremental tap decode) ----
        {
            int kk = k0 + kstart;
            int tap = kk / cin;
            int c = kk - tap * cin;
            bool valid;
            long base = conv_base(tap, fox, foy, foz, d_in, ks, stride, pad, cin, mode, valid);
            for (int j = 0; j < 16; ++j) {
                if (c == cin) {
                    ++tap; c = 0;
                    base = conv_base(tap, fox, foy, foz, d_in, ks, stride, pad, cin, mode, valid);
                }
                float av = (valid && (kk + j) < K) ? in[base + c] : 0.0f;
                lds_a[frow * AROWS + kstart + j] = (_Float16)av;
                ++c;
            }
        }
#if USE_TDM
        if (wave == 0) {
            if (s + 1 < nslab) {   // overlap next weight DMA with this compute
                tdm_load_b_slab(whT + (long)(k0 + 32),
                                (unsigned)(unsigned long long)(void*)&lds_b[(s + 1) & 1][0],
                                Kp, cout);
                __builtin_amdgcn_s_wait_tensorcnt(1);  // in-order: slab s landed
            } else {
                __builtin_amdgcn_s_wait_tensorcnt(0);
            }
        }
#endif
        __syncthreads();
        // ---- A fragment: two ds_read_b128 ----
        v16h a;
        {
            const v8h lo = *(const v8h*)&lds_a[arow * AROWS + khalf];
            const v8h hi = *(const v8h*)&lds_a[arow * AROWS + 16 + khalf];
#pragma unroll
            for (int e = 0; e < 8; ++e) { a[e] = lo[e]; a[8 + e] = hi[e]; }
        }
        // ---- B fragments + WMMA per n-tile ----
#pragma unroll
        for (int nt = 0; nt < NT; ++nt) {
            v16h b;
#if USE_TDM
            const _Float16* bp = &lds_b[s & 1][(nt * 16 + bcol) * 32];
            const v8h blo = *(const v8h*)(bp + khalf);
            const v8h bhi = *(const v8h*)(bp + 16 + khalf);
#else
            const _Float16* bp = whT + (long)(nt * 16 + bcol) * Kp + k0;
            __builtin_prefetch(bp + 32, 0, 1);
            const v8h blo = *(const v8h*)(bp + khalf);
            const v8h bhi = *(const v8h*)(bp + 16 + khalf);
#endif
#pragma unroll
            for (int e = 0; e < 8; ++e) { b[e] = blo[e]; b[8 + e] = bhi[e]; }
            acc[nt] = __builtin_amdgcn_wmma_f32_16x16x32_f16(
                false, a, false, b, (short)0, acc[nt], false, false);
        }
        __syncthreads();
    }
    // ---- epilogue: C/D layout lane->(M = r + 8*(lane>>4), N = lane&15) ----
#pragma unroll
    for (int nt = 0; nt < NT; ++nt) {
#pragma unroll
        for (int r = 0; r < 8; ++r) {
            long m = mblk + wave * 16 + r + (long)((lane >> 4) * 8);
            int n = nt * 16 + bcol;
            if (m < Mtot && n < cout) out[m * cout + n] = acc[nt][r];
        }
    }
}

// ---------------------------------------------------------------------------
// Plain row-major GEMM (point transforms): A[M,K] f32 (aligned float4 loads)
// x whT[cout][Kp] f16 (b128 loads).
// ---------------------------------------------------------------------------
template <int NT>
__global__ __launch_bounds__(256)
void k_gemm_wmma(const float* __restrict__ A, const _Float16* __restrict__ whT,
                 float* __restrict__ out, long M, int K, int Kp, int cout) {
    const int lane = threadIdx.x & 31;
    const int wave = threadIdx.x >> 5;
    const long m0 = (long)blockIdx.x * 128 + (long)wave * 16;
    if (m0 >= M) return;  // wave-uniform

    long mA = m0 + (lane & 15); if (mA >= M) mA = M - 1;
    const int khalf = (lane >> 4) * 8;
    const int bcol  = lane & 15;
    const float* ar = A + mA * K;

    v8f acc[NT];
#pragma unroll
    for (int nt = 0; nt < NT; ++nt) acc[nt] = (v8f){};

    for (int k0 = 0; k0 < Kp; k0 += 32) {
        v16h a = {};
        if (k0 + khalf < K) {  // chunk fully in/out (K multiple of 8)
            const v4f p0 = *(const v4f*)(ar + k0 + khalf);
            const v4f p1 = *(const v4f*)(ar + k0 + khalf + 4);
#pragma unroll
            for (int e = 0; e < 4; ++e) { a[e] = (_Float16)p0[e]; a[4 + e] = (_Float16)p1[e]; }
        }
        if (k0 + 16 + khalf < K) {
            const v4f p0 = *(const v4f*)(ar + k0 + 16 + khalf);
            const v4f p1 = *(const v4f*)(ar + k0 + 16 + khalf + 4);
#pragma unroll
            for (int e = 0; e < 4; ++e) { a[8 + e] = (_Float16)p0[e]; a[12 + e] = (_Float16)p1[e]; }
        }
#pragma unroll
        for (int nt = 0; nt < NT; ++nt) {
            const _Float16* bp = whT + (long)(nt * 16 + bcol) * Kp + k0;
            v16h b;
            const v8h blo = *(const v8h*)(bp + khalf);
            const v8h bhi = *(const v8h*)(bp + 16 + khalf);
#pragma unroll
            for (int e = 0; e < 8; ++e) { b[e] = blo[e]; b[8 + e] = bhi[e]; }
            acc[nt] = __builtin_amdgcn_wmma_f32_16x16x32_f16(
                false, a, false, b, (short)0, acc[nt], false, false);
        }
    }
#pragma unroll
    for (int nt = 0; nt < NT; ++nt) {
#pragma unroll
        for (int r = 0; r < 8; ++r) {
            long m = m0 + r + (long)((lane >> 4) * 8);
            int n = nt * 16 + bcol;
            if (m < M && n < cout) out[m * cout + n] = acc[nt][r];
        }
    }
}

// ---------------------------------------------------------------------------
// Host orchestration
// ---------------------------------------------------------------------------
extern "C" void kernel_launch(void* const* d_in, const int* in_sizes, int n_in,
                              void* d_out, int out_size, void* d_ws, size_t ws_size,
                              hipStream_t stream) {
    (void)n_in; (void)out_size; (void)ws_size;

    enum {
        IC_COORDS = 0, IC_FEATS,
        PT0_B, PT0_BIAS, PT0_G, PT0_W,
        PT1_B, PT1_BIAS, PT1_G, PT1_W,
        S1D_B, S1D_G, S1D_W,
        S1R1C1_B, S1R1C1_G, S1R1C1_W, S1R1C2_B, S1R1C2_G, S1R1C2_W,
        S1R1D_B, S1R1D_G, S1R1D_W,
        S1R2C1_B, S1R2C1_G, S1R2C1_W, S1R2C2_B, S1R2C2_G, S1R2C2_W,
        S2D_B, S2D_G, S2D_W,
        S2R1C1_B, S2R1C1_G, S2R1C1_W, S2R1C2_B, S2R1C2_G, S2R1C2_W,
        S2R1D_B, S2R1D_G, S2R1D_W,
        S2R2C1_B, S2R2C1_G, S2R2C1_W, S2R2C2_B, S2R2C2_G, S2R2C2_W,
        STEM_B, STEM_G, STEM_W,
        U1DE_B, U1DE_G, U1DE_W,
        U1R1C1_B, U1R1C1_G, U1R1C1_W, U1R1C2_B, U1R1C2_G, U1R1C2_W,
        U1R1D_B, U1R1D_G, U1R1D_W,
        U1R2C1_B, U1R2C1_G, U1R2C1_W, U1R2C2_B, U1R2C2_G, U1R2C2_W,
        U2DE_B, U2DE_G, U2DE_W,
        U2R1C1_B, U2R1C1_G, U2R1C1_W, U2R1C2_B, U2R1C2_G, U2R1C2_W,
        U2R1D_B, U2R1D_G, U2R1D_W,
        U2R2C1_B, U2R2C1_G, U2R2C1_W, U2R2C2_B, U2R2C2_G, U2R2C2_W
    };

    const float* coords = (const float*)d_in[IC_COORDS];
    const float* feats  = (const float*)d_in[IC_FEATS];
    const int N = in_sizes[IC_COORDS] / 3;

    const int D1 = 48, D2 = 24, D3 = 12;
    const long M1 = (long)D1 * D1 * D1;
    const long M2 = (long)D2 * D2 * D2;
    const long M3 = (long)D3 * D3 * D3;

    char* ws = (char*)d_ws;
    size_t off = 0;
    auto allocf = [&](size_t n) -> float* {
        float* p = (float*)(ws + off);
        off += ((n * sizeof(float) + 255) / 256) * 256;
        return p;
    };
    auto alloch = [&](size_t n) -> _Float16* {
        _Float16* p = (_Float16*)(ws + off);
        off += ((n * sizeof(_Float16) + 255) / 256) * 256;
        return p;
    };

    float* x0 = allocf(M1 * 16);
    float* x1 = allocf(M2 * 32);
    float* x2 = allocf(M3 * 64);
    float* z0 = allocf((size_t)N * 16);
    float* z1 = allocf((size_t)N * 64);
    float* ppre = allocf((size_t)N * 64);
    float* t0 = allocf(M1 * 96);
    float* t1 = allocf(M1 * 96);
    float* t2 = allocf(M1 * 96);
    float* t3 = allocf(M1 * 96);
    float* cnt = allocf(M1);
    float* meanv = allocf(128);
    float* varv  = allocf(128);
    _Float16* whT = alloch(64 * 2304);  // [cout][Kp], max cout=64 / Kp=2176

    const int TPB = 256;
    auto nblk = [](long tot) -> unsigned {
        long b = (tot + 255) / 256;
        if (b > 65536) b = 65536;
        if (b < 1) b = 1;
        return (unsigned)b;
    };

    auto voxelize = [&](const float* f, int C, float strideF, int d, float* grid) {
        long M = (long)d * d * d;
        k_zero<<<nblk(M * C), TPB, 0, stream>>>(grid, M * C);
        k_zero<<<nblk(M), TPB, 0, stream>>>(cnt, M);
        k_scatter<<<nblk((long)N * C), TPB, 0, stream>>>(f, coords, grid, cnt, N, C, strideF, d);
        k_voxnorm<<<nblk(M * C), TPB, 0, stream>>>(grid, cnt, M, C);
    };

    auto conv_bn = [&](const float* inb, int din, int dout, int cin, int cout,
                       int ks, int stride, int pad, int mode,
                       int ib, int ig, int iw, float* outb, float* pre,
                       const float* res, int relu) {
        int K = mode ? 8 * cin : ks * ks * ks * cin;
        int Kp = ((K + 31) / 32) * 32;
        long Mtot = (long)dout * dout * dout;
        k_w2hT<<<nblk((long)cout * Kp), TPB, 0, stream>>>(
            (const float*)d_in[iw], whT, K, Kp, cout);
        dim3 g((unsigned)((Mtot + 127) / 128));
        switch (cout / 16) {
        case 1: k_conv_wmma<1><<<g, TPB, 0, stream>>>(inb, whT, pre, din, dout, cin, cout, K, Kp, ks, stride, pad, mode); break;
        case 2: k_conv_wmma<2><<<g, TPB, 0, stream>>>(inb, whT, pre, din, dout, cin, cout, K, Kp, ks, stride, pad, mode); break;
        case 3: k_conv_wmma<3><<<g, TPB, 0, stream>>>(inb, whT, pre, din, dout, cin, cout, K, Kp, ks, stride, pad, mode); break;
        default: k_conv_wmma<4><<<g, TPB, 0, stream>>>(inb, whT, pre, din, dout, cin, cout, K, Kp, ks, stride, pad, mode); break;
        }
        k_bn_stats<<<cout, TPB, 0, stream>>>(pre, Mtot, cout, meanv, varv);
        k_bn_apply<<<nblk(Mtot * cout), TPB, 0, stream>>>(
            pre, meanv, varv, (const float*)d_in[ig], (const float*)d_in[ib],
            res, outb, Mtot, cout, relu);
    };

    auto resblock = [&](const float* inb, int d, int cin, int cout,
                        int c1b, int c1g, int c1w, int c2b, int c2g, int c2w,
                        int db, int dg, int dw, bool hasD, float* outb) {
        conv_bn(inb, d, d, cin, cout, 3, 1, 1, 0, c1b, c1g, c1w, t1, t0, nullptr, 1);
        const float* sc = inb;
        if (hasD) {
            conv_bn(inb, d, d, cin, cout, 1, 1, 0, 0, db, dg, dw, t2, t0, nullptr, 0);
            sc = t2;
        }
        conv_bn(t1, d, d, cout, cout, 3, 1, 1, 0, c2b, c2g, c2w, outb, t0, sc, 1);
    };

    auto point_transform = [&](const float* A, int K, int cout,
                               int ib, int ig, int iw, float* outb) {
        int Kp = ((K + 31) / 32) * 32;
        k_w2hT<<<nblk((long)cout * Kp), TPB, 0, stream>>>(
            (const float*)d_in[iw], whT, K, Kp, cout);
        dim3 g((unsigned)(((long)N + 127) / 128));
        if (cout / 16 == 3)
            k_gemm_wmma<3><<<g, TPB, 0, stream>>>(A, whT, ppre, (long)N, K, Kp, cout);
        else
            k_gemm_wmma<4><<<g, TPB, 0, stream>>>(A, whT, ppre, (long)N, K, Kp, cout);
        k_bn_stats<<<cout, TPB, 0, stream>>>(ppre, (long)N, cout, meanv, varv);
        k_bn_apply<<<nblk((long)N * cout), TPB, 0, stream>>>(
            ppre, meanv, varv, (const float*)d_in[ig], (const float*)d_in[ib],
            nullptr, outb, (long)N, cout, 1);
    };

    // --- forward pass -----------------------------------------------------
    voxelize(feats, 8, 1.0f, D1, t1);
    conv_bn(t1, D1, D1, 8, 16, 3, 1, 1, 0, STEM_B, STEM_G, STEM_W, x0, t0, nullptr, 1);
    k_devox<<<nblk((long)N * 16), TPB, 0, stream>>>(x0, coords, z0, N, 16, 1.0f, D1);

    voxelize(z0, 16, 1.0f, D1, t1);
    conv_bn(t1, D1, D2, 16, 16, 2, 2, 0, 0, S1D_B, S1D_G, S1D_W, t2, t0, nullptr, 1);
    resblock(t2, D2, 16, 32, S1R1C1_B, S1R1C1_G, S1R1C1_W,
             S1R1C2_B, S1R1C2_G, S1R1C2_W, S1R1D_B, S1R1D_G, S1R1D_W, true, t3);
    resblock(t3, D2, 32, 32, S1R2C1_B, S1R2C1_G, S1R2C1_W,
             S1R2C2_B, S1R2C2_G, S1R2C2_W, 0, 0, 0, false, x1);

    conv_bn(x1, D2, D3, 32, 32, 2, 2, 0, 0, S2D_B, S2D_G, S2D_W, t2, t0, nullptr, 1);
    resblock(t2, D3, 32, 64, S2R1C1_B, S2R1C1_G, S2R1C1_W,
             S2R1C2_B, S2R1C2_G, S2R1C2_W, S2R1D_B, S2R1D_G, S2R1D_W, true, t3);
    resblock(t3, D3, 64, 64, S2R2C1_B, S2R2C1_G, S2R2C1_W,
             S2R2C2_B, S2R2C2_G, S2R2C2_W, 0, 0, 0, false, x2);

    k_devox<<<nblk((long)N * 64), TPB, 0, stream>>>(x2, coords, z1, N, 64, 4.0f, D3);
    point_transform(z0, 16, 64, PT0_B, PT0_G, PT0_W, ppre);
    k_add_inplace<<<nblk((long)N * 64), TPB, 0, stream>>>(z1, ppre, (long)N * 64);

    voxelize(z1, 64, 4.0f, D3, t1);
    conv_bn(t1, D3, D2, 64, 48, 2, 2, 0, 1, U1DE_B, U1DE_G, U1DE_W, t2, t0, nullptr, 1);
    k_concat<<<nblk(M2 * 80), TPB, 0, stream>>>(t2, 48, x1, 32, t3, M2);
    resblock(t3, D2, 80, 48, U1R1C1_B, U1R1C1_G, U1R1C1_W,
             U1R1C2_B, U1R1C2_G, U1R1C2_W, U1R1D_B, U1R1D_G, U1R1D_W, true, t2);
    resblock(t2, D2, 48, 48, U1R2C1_B, U1R2C1_G, U1R2C1_W,
             U1R2C2_B, U1R2C2_G, U1R2C2_W, 0, 0, 0, false, t3);

    conv_bn(t3, D2, D1, 48, 48, 2, 2, 0, 1, U2DE_B, U2DE_G, U2DE_W, t1, t0, nullptr, 1);
    k_concat<<<nblk(M1 * 64), TPB, 0, stream>>>(t1, 48, x0, 16, t2, M1);
    resblock(t2, D1, 64, 48, U2R1C1_B, U2R1C1_G, U2R1C1_W,
             U2R1C2_B, U2R1C2_G, U2R1C2_W, U2R1D_B, U2R1D_G, U2R1D_W, true, t3);
    resblock(t3, D1, 48, 48, U2R2C1_B, U2R2C1_G, U2R2C1_W,
             U2R2C2_B, U2R2C2_G, U2R2C2_W, 0, 0, 0, false, t1);

    float* outp = (float*)d_out;
    k_devox<<<nblk((long)N * 48), TPB, 0, stream>>>(t1, coords, outp, N, 48, 1.0f, D1);
    point_transform(z1, 64, 48, PT1_B, PT1_G, PT1_W, ppre);
    k_add_inplace<<<nblk((long)N * 48), TPB, 0, stream>>>(outp, ppre, (long)N * 48);
}
